// AttentionCircuit_25829933318716
// MI455X (gfx1250) — matrix operand order repaired and checked by reading, not compiled, not run on hardware
//
#include <hip/hip_runtime.h>
#include <math.h>

#define DEV __device__ __forceinline__

typedef __bf16 bf16;
typedef __attribute__((ext_vector_type(16))) __bf16 v16bf;
typedef __attribute__((ext_vector_type(8)))  float  v8f;

static constexpr int Bc  = 2;
static constexpr int Sc  = 2048;
static constexpr int Dc  = 1024;
static constexpr int Hc  = 16;
static constexpr int DHc = 64;
static constexpr int TOK = Bc * Sc;          // 4096 tokens
static constexpr int NRc = 16;               // router experts
static constexpr int Rc  = 16;               // rank
static constexpr int CDIM = NRc * Rc;        // 256

// ---------------------------------------------------------------- helpers
union U32B2 { unsigned int u; bf16 h[2]; };

DEV bf16 f2bf(float f) {
  union { float f; unsigned int u; } c; c.f = f;
  unsigned int r = c.u + 0x7FFFu + ((c.u >> 16) & 1u);   // round-nearest-even
  union { unsigned short s; bf16 b; } o;
  o.s = (unsigned short)(r >> 16);
  return o.b;
}

DEV v8f vzero8() {
  v8f z;
#pragma unroll
  for (int i = 0; i < 8; ++i) z[i] = 0.0f;
  return z;
}

DEV float rmax16(float x) {
  x = fmaxf(x, __shfl_xor(x, 1, 32));
  x = fmaxf(x, __shfl_xor(x, 2, 32));
  x = fmaxf(x, __shfl_xor(x, 4, 32));
  x = fmaxf(x, __shfl_xor(x, 8, 32));
  return x;
}
DEV float rsum16(float x) {
  x += __shfl_xor(x, 1, 32);
  x += __shfl_xor(x, 2, 32);
  x += __shfl_xor(x, 4, 32);
  x += __shfl_xor(x, 8, 32);
  return x;
}

DEV v8f wmma_bf16(v16bf a, v16bf b, v8f c) {
  return __builtin_amdgcn_wmma_f32_16x16x32_bf16(false, a, false, b, (short)0, c,
                                                 false, false);
}

// A-matrix (16x32 bf16) fragment: lane L holds row m=L%16, K = 16*(L/16) + 0..15 (contiguous).
DEV v16bf load_afrag(const bf16* row, int kbase, int g) {
  v16bf f;
#pragma unroll
  for (int v = 0; v < 8; ++v) {
    int k = kbase + 2 * v + 16 * g;
    U32B2 cv;
    cv.u = *reinterpret_cast<const unsigned int*>(row + k);
    f[2 * v]     = cv.h[0];
    f[2 * v + 1] = cv.h[1];
  }
  return f;
}

DEV unsigned lds_addr32(const void* p) {      // generic shared ptr -> LDS byte addr
  return (unsigned)(unsigned long long)(uintptr_t)p;
}

// Issue one async 16B global->LDS copy (tracked by ASYNCcnt).
DEV void async_cp16(const bf16* g, const bf16* l) {
  asm volatile("global_load_async_to_lds_b128 %0, %1, off"
               :: "v"(lds_addr32(l)), "v"((unsigned long long)(uintptr_t)g)
               : "memory");
}

// ---------------------------------------------------------------- kernels

__global__ void ac_zero_acc(float* acc) {
  if (threadIdx.x < 48) acc[threadIdx.x] = 0.0f;
}

// x (f32, 4096x1024) -> xb (bf16)
__global__ void ac_conv_x(const float* __restrict__ x, bf16* __restrict__ xb) {
  int i = blockIdx.x * blockDim.x + threadIdx.x;
  if (i < TOK * Dc) xb[i] = f2bf(x[i]);
}

// Pack all static weights into bf16, B-transposed layouts.
__global__ void ac_conv_w(const float* __restrict__ f_neurons,
                          const float* __restrict__ r_neurons,
                          const float* __restrict__ W_O,
                          bf16* __restrict__ fbt,    // [1024 cols][1024 k]
                          bf16* __restrict__ rqk_t,  // [1024 d][256 nr]
                          bf16* __restrict__ rv_t,   // [1024 d][256 nr]
                          bf16* __restrict__ wo_t) { // [1024 cols][1024 k]
  int i = blockIdx.x * blockDim.x + threadIdx.x;
  if (i < 1048576) {
    int nc = i >> 10, k = i & 1023;
    fbt[i] = f2bf(f_neurons[(size_t)(nc >> 4) * 16384 + (size_t)k * 16 + (nc & 15)]);
    return;
  }
  int j = i - 1048576;
  if (j < 262144) {
    int d = j >> 8, nr = j & 255;
    rqk_t[j] = f2bf(r_neurons[(size_t)nr * 1024 + d]);
    return;
  }
  j -= 262144;
  if (j < 262144) {
    int d = j >> 8, nr = j & 255;
    rv_t[j] = f2bf(r_neurons[(size_t)(256 + nr) * 1024 + d]);
    return;
  }
  j -= 262144;
  if (j < 1048576) {
    int c = j >> 10, k = j & 1023;
    wo_t[j] = f2bf(W_O[(size_t)k * 1024 + c]);
  }
}

// bf16 WMMA GEMM, 32x32 macro-tile per wave (2x2 WMMA tiles, 4 wmma / K-step).
// C(f32 MxN) = A(bf16 MxK) * Bt(bf16 NxK, B transposed). 8 waves/block.
__global__ __launch_bounds__(256)
void ac_gemm_bf16(const bf16* __restrict__ A, const bf16* __restrict__ Bt,
                  float* __restrict__ C, int M, int N, int K) {
  int lane = threadIdx.x & 31;
  int wid  = threadIdx.x >> 5;
  int n0 = blockIdx.x * 256 + wid * 32;
  int m0 = blockIdx.y * 32;
  if (n0 >= N || m0 >= M) return;
  int g  = lane >> 4;
  int lm = lane & 15;

  const bf16* arow0 = A  + (size_t)(m0 + lm) * K;
  const bf16* arow1 = arow0 + (size_t)16 * K;
  const bf16* bcol0 = Bt + (size_t)(n0 + lm) * K;
  const bf16* bcol1 = bcol0 + (size_t)16 * K;

  v8f acc00 = vzero8(), acc01 = vzero8(), acc10 = vzero8(), acc11 = vzero8();
  for (int kb = 0; kb < K; kb += 32) {
    __builtin_prefetch(arow0 + kb + 256, 0, 1);   // global_prefetch_b8
    __builtin_prefetch(bcol0 + kb + 256, 0, 1);
    v16bf a0 = load_afrag(arow0, kb, g);
    v16bf a1 = load_afrag(arow1, kb, g);
    v16bf b0 = *reinterpret_cast<const v16bf*>(bcol0 + kb + 16 * g);
    v16bf b1 = *reinterpret_cast<const v16bf*>(bcol1 + kb + 16 * g);
    acc00 = wmma_bf16(a0, b0, acc00);
    acc01 = wmma_bf16(a0, b1, acc01);
    acc10 = wmma_bf16(a1, b0, acc10);
    acc11 = wmma_bf16(a1, b1, acc11);
  }
#pragma unroll
  for (int j = 0; j < 8; ++j) {
    size_t r0 = (size_t)(m0 + j + 8 * g) * N;
    size_t r1 = (size_t)(m0 + 16 + j + 8 * g) * N;
    C[r0 + n0 + lm]      = acc00[j];
    C[r0 + n0 + 16 + lm] = acc01[j];
    C[r1 + n0 + lm]      = acc10[j];
    C[r1 + n0 + 16 + lm] = acc11[j];
  }
}

// Per-token router + C-tensor construction. One block (256 thr) per (b,s) token.
__global__ __launch_bounds__(256)
void ac_router(const float* __restrict__ allh,
               const float* __restrict__ fqk_wQ, const float* __restrict__ fqk_wK,
               const float* __restrict__ fv_w,
               const float* __restrict__ fqk_emb, const float* __restrict__ fv_emb,
               const float* __restrict__ Wp_qk, const float* __restrict__ bp_qk,
               const float* __restrict__ Wp_v,  const float* __restrict__ bp_v,
               const float* __restrict__ Wr_qk, const float* __restrict__ Wr_v,
               bf16* __restrict__ Cq, bf16* __restrict__ Ck, bf16* __restrict__ Cv,
               float* __restrict__ meanacc) {
  __shared__ float sh_ah[1024];
  __shared__ float sh_w[3][32];
  __shared__ float sh_h[3][16];
  __shared__ float sh_in[3][128];
  __shared__ float sh_sm[3][16];
  __shared__ float sh_C[3][256];

  int tok = blockIdx.x;
  int tid = threadIdx.x;
  int b = tok >> 11, s = tok & 2047;

  for (int i = tid; i < 1024; i += 256) sh_ah[i] = allh[(size_t)tok * 1024 + i];
  for (int i = tid; i < 768; i += 256) (&sh_C[0][0])[i] = 0.0f;
  __syncthreads();

  for (int p = 0; p < 2; ++p) {
    size_t gbase = (((size_t)p * Bc + b) * Sc + s) * 32;
    if (tid < 96) {
      int st = tid >> 5, n = tid & 31;
      const float* src = (st == 0) ? fqk_wQ : (st == 1) ? fqk_wK : fv_w;
      sh_w[st][n] = src[gbase + n];
    }
    __syncthreads();

    if (tid < 48) {                       // h[st][r] = sum_n w[n] * allh[n,r]
      int st = tid >> 4, r = tid & 15;
      const float* ah = (st == 2) ? (sh_ah + 512) : sh_ah;
      float a = 0.0f;
      for (int n = 0; n < 32; ++n) a += sh_w[st][n] * ah[n * 16 + r];
      sh_h[st][r] = a;
    }
    __syncthreads();

    for (int i = tid; i < 384; i += 256) { // router inputs (proj || ctx)
      int st = i >> 7, j = i & 127;
      float v;
      if (j < 64) {
        const float* Wp = (st == 2) ? Wp_v : Wp_qk;
        const float* bp = (st == 2) ? bp_v : bp_qk;
        float a = 0.0f;
        for (int r = 0; r < 16; ++r) a += sh_h[st][r] * Wp[r * 64 + j];
        v = a + bp[j];
      } else {
        const float* emb = (st == 2) ? fv_emb : fqk_emb;
        int d = j - 64;
        float a = 0.0f;
        for (int n = 0; n < 32; ++n) a += sh_w[st][n] * emb[n * 64 + d];
        v = a;
      }
      sh_in[st][j] = v;
    }
    __syncthreads();

    if (tid < 48) {                       // logits
      int st = tid >> 4, o = tid & 15;
      const float* Wr = (st == 2) ? Wr_v : Wr_qk;
      float a = 0.0f;
      for (int j = 0; j < 128; ++j) a += sh_in[st][j] * Wr[j * 16 + o];
      sh_sm[st][o] = a;
    }
    __syncthreads();

    if (tid < 3) {                        // softmax (in place)
      int st = tid;
      float e[16];
      float mx = -3.0e38f;
      for (int o = 0; o < 16; ++o) mx = fmaxf(mx, sh_sm[st][o]);
      float sm = 0.0f;
      for (int o = 0; o < 16; ++o) { e[o] = __expf(sh_sm[st][o] - mx); sm += e[o]; }
      float inv = 1.0f / sm;
      for (int o = 0; o < 16; ++o) sh_sm[st][o] = e[o] * inv;
    }
    __syncthreads();

    if (tid < 48) {
      int st = tid >> 4, o = tid & 15;
      atomicAdd(&meanacc[st * 16 + o], sh_sm[st][o]);
    }
    {                                     // C[nr] += w[n] * h[r], nr = n*16+r
      int n = tid >> 4, r = tid & 15;
      for (int st = 0; st < 3; ++st) sh_C[st][tid] += sh_sm[st][n] * sh_h[st][r];
    }
    __syncthreads();
  }

  Cq[(size_t)tok * CDIM + tid] = f2bf(sh_C[0][tid]);
  Ck[(size_t)tok * CDIM + tid] = f2bf(sh_C[1][tid]);
  Cv[(size_t)tok * CDIM + tid] = f2bf(sh_C[2][tid]);
}

// Per-token ||Q|| -> scale
__global__ __launch_bounds__(256)
void ac_qscale(const float* __restrict__ Qf, float* __restrict__ scale) {
  __shared__ float red[256];
  int tok = blockIdx.x, tid = threadIdx.x;
  float a = 0.0f;
  for (int i = tid; i < Dc; i += 256) {
    float q = Qf[(size_t)tok * Dc + i];
    a += q * q;
  }
  red[tid] = a;
  __syncthreads();
  for (int s2 = 128; s2 > 0; s2 >>= 1) {
    if (tid < s2) red[tid] += red[tid + s2];
    __syncthreads();
  }
  if (tid == 0) {
    float norm = sqrtf(red[0]);
    scale[tok] = (norm > 1e-6f) ? 1.0f : norm * 1e-6f;
  }
}

// Token-major f32 Q/K/V -> head-major bf16 (Q,K row-major; V transposed [dh][s]).
__global__ void ac_pack_qkv(const float* __restrict__ Qf, const float* __restrict__ Kf,
                            const float* __restrict__ Vf,
                            bf16* __restrict__ Qb, bf16* __restrict__ Kb,
                            bf16* __restrict__ Vt) {
  int idx = blockIdx.x * blockDim.x + threadIdx.x;
  if (idx >= TOK * Dc) return;
  int tok = idx >> 10, d = idx & 1023;
  int b = tok >> 11, s = tok & 2047;
  int h = d >> 6, dh = d & 63;
  size_t bh = (size_t)b * Hc + h;
  Qb[(bh * Sc + s) * DHc + dh] = f2bf(Qf[idx]);
  Kb[(bh * Sc + s) * DHc + dh] = f2bf(Kf[idx]);
  Vt[(bh * DHc + dh) * Sc + s] = f2bf(Vf[idx]);
}

// Stage one 32-key K tile (32x64) and V tile (64x32) into LDS via async copies.
// 16 x b128 per wave => ASYNCcnt += 16.
DEV void stage_kv(const bf16* Kg, const bf16* Vtg, bf16* dK, bf16* dV,
                  int k0, int lane) {
#pragma unroll
  for (int t = 0; t < 8; ++t) {           // K: rows of 64 dh (128B) x 32 keys
    int c = lane + 32 * t;                // 16B chunk id
    int key = c >> 3, o = c & 7;
    async_cp16(Kg + (size_t)(k0 + key) * DHc + o * 8, dK + key * DHc + o * 8);
  }
#pragma unroll
  for (int t = 0; t < 8; ++t) {           // V: 64 dh rows of 32 keys (64B)
    int c = lane + 32 * t;
    int dh = c >> 2, seg = c & 3;
    async_cp16(Vtg + (size_t)dh * Sc + k0 + seg * 8, dV + dh * 32 + seg * 8);
  }
}

// Causal flash attention: one wave per 16-row query tile; async double-buffered
// K/V staging in LDS; bf16 WMMA for QK^T and PV. 2 waves / block.
__global__ __launch_bounds__(64)
void ac_attn(const bf16* __restrict__ Qb, const bf16* __restrict__ Kb,
             const bf16* __restrict__ Vt, float* __restrict__ Oout) {
  __shared__ bf16 stK[2][2][32 * 64];     // [wave][buf][key][dh]
  __shared__ bf16 stV[2][2][64 * 32];     // [wave][buf][dh][key]
  __shared__ bf16 ldsP[2][16 * 32];       // per-wave P staging

  int lane = threadIdx.x & 31;
  int wid  = threadIdx.x >> 5;
  int bh = blockIdx.y;
  int qt = blockIdx.x * 2 + wid;
  int q0 = qt * 16;
  int g = lane >> 4, lm = lane & 15;

  const bf16* Kg  = Kb + (size_t)bh * Sc * DHc;
  const bf16* Vtg = Vt + (size_t)bh * DHc * Sc;

  const bf16* qrow = Qb + ((size_t)bh * Sc + q0 + lm) * DHc;
  v16bf qf[2];
#pragma unroll
  for (int ks = 0; ks < 2; ++ks) qf[ks] = load_afrag(qrow, ks * 32, g);

  v8f Oacc[4];
#pragma unroll
  for (int nb = 0; nb < 4; ++nb) Oacc[nb] = vzero8();
  float mi[8], li[8];
#pragma unroll
  for (int j = 0; j < 8; ++j) { mi[j] = -3.0e38f; li[j] = 0.0f; }

  bf16* lp = ldsP[wid];
  int nkb = (q0 + 47) >> 5;               // 32-key blocks covering keys <= q0+15

  stage_kv(Kg, Vtg, stK[wid][0], stV[wid][0], 0, lane);

  for (int kb = 0; kb < nkb; ++kb) {
    int cur = kb & 1;
    if (kb + 1 < nkb) {                   // prefetch next block, then wait for cur
      stage_kv(Kg, Vtg, stK[wid][cur ^ 1], stV[wid][cur ^ 1], (kb + 1) * 32, lane);
      asm volatile("s_wait_asynccnt 0x10" ::: "memory");
    } else {
      asm volatile("s_wait_asynccnt 0x0" ::: "memory");
    }
    int k0 = kb * 32;
    const bf16* sk = stK[wid][cur];
    const bf16* sv = stV[wid][cur];

    v8f st[2];
#pragma unroll
    for (int sbt = 0; sbt < 2; ++sbt) {
      v8f sacc = vzero8();
#pragma unroll
      for (int ks = 0; ks < 2; ++ks) {
        v16bf kf = *reinterpret_cast<const v16bf*>(
            sk + (sbt * 16 + lm) * DHc + ks * 32 + 16 * g);
        sacc = wmma_bf16(qf[ks], kf, sacc);
      }
      st[sbt] = sacc;
    }

    float alpha[8];
#pragma unroll
    for (int j = 0; j < 8; ++j) {
      int row = q0 + j + 8 * g;
      float s0 = (k0 + lm      <= row) ? st[0][j] * 0.125f : -3.0e38f;
      float s1 = (k0 + 16 + lm <= row) ? st[1][j] * 0.125f : -3.0e38f;
      float mx = rmax16(fmaxf(s0, s1));
      float mn = fmaxf(mi[j], mx);
      float a  = __expf(mi[j] - mn);
      float p0 = __expf(s0 - mn);
      float p1 = __expf(s1 - mn);
      li[j] = li[j] * a + rsum16(p0 + p1);
      mi[j] = mn;
      alpha[j] = a;
      st[0][j] = p0;
      st[1][j] = p1;
    }

    // stage P (16x32 bf16) in LDS, then reload in A-fragment order.
#pragma unroll
    for (int j = 0; j < 8; ++j) {
      lp[(j + 8 * g) * 32 + lm]      = f2bf(st[0][j]);
      lp[(j + 8 * g) * 32 + 16 + lm] = f2bf(st[1][j]);
    }
    asm volatile("s_wait_dscnt 0" ::: "memory");   // DS in-order within wave
    v16bf pf;
#pragma unroll
    for (int v = 0; v < 8; ++v) {
      int k = 2 * v + 16 * g;
      pf[2 * v]     = lp[lm * 32 + k];
      pf[2 * v + 1] = lp[lm * 32 + k + 1];
    }
    asm volatile("s_wait_dscnt 0" ::: "memory");

#pragma unroll
    for (int nb = 0; nb < 4; ++nb) {
#pragma unroll
      for (int j = 0; j < 8; ++j) Oacc[nb][j] *= alpha[j];
      v16bf vf = *reinterpret_cast<const v16bf*>(
          sv + (nb * 16 + lm) * 32 + 16 * g);
      Oacc[nb] = wmma_bf16(pf, vf, Oacc[nb]);
    }
  }

  int b = bh >> 4, h = bh & 15;
#pragma unroll
  for (int nb = 0; nb < 4; ++nb)
#pragma unroll
    for (int j = 0; j < 8; ++j) {
      float o = Oacc[nb][j] / li[j];
      Oout[((size_t)b * Sc + q0 + j + 8 * g) * Dc + h * DHc + nb * 16 + lm] = o;
    }
}

// attn_out * scale -> bf16
__global__ void ac_scaleconv(const float* __restrict__ ao, const float* __restrict__ scale,
                             bf16* __restrict__ aso) {
  int idx = blockIdx.x * blockDim.x + threadIdx.x;
  if (idx >= TOK * Dc) return;
  aso[idx] = f2bf(ao[idx] * scale[idx >> 10]);
}

// aux = sum over 3 routers of N * sum(mean_p^2)
__global__ void ac_aux(const float* __restrict__ meanacc, float* __restrict__ out) {
  float aux = 0.0f;
  for (int st = 0; st < 3; ++st) {
    float s2 = 0.0f;
    for (int o = 0; o < 16; ++o) {
      float m = meanacc[st * 16 + o] * (1.0f / 8192.0f);
      s2 += m * m;
    }
    aux += 16.0f * s2;
  }
  out[(size_t)TOK * Dc] = aux;
}

// ---------------------------------------------------------------- launch
extern "C" void kernel_launch(void* const* d_in, const int* in_sizes, int n_in,
                              void* d_out, int out_size, void* d_ws, size_t ws_size,
                              hipStream_t stream) {
  (void)in_sizes; (void)n_in; (void)out_size; (void)ws_size;
  const float* x        = (const float*)d_in[0];
  const float* fqk_wQ   = (const float*)d_in[1];
  const float* fqk_wK   = (const float*)d_in[2];
  const float* fv_w     = (const float*)d_in[3];
  const float* f_neurons= (const float*)d_in[4];
  const float* r_neurons= (const float*)d_in[5];
  const float* fqk_emb  = (const float*)d_in[6];
  const float* fv_emb   = (const float*)d_in[7];
  const float* Wp_qk    = (const float*)d_in[8];
  const float* bp_qk    = (const float*)d_in[9];
  const float* Wp_v     = (const float*)d_in[10];
  const float* bp_v     = (const float*)d_in[11];
  const float* Wr_qk    = (const float*)d_in[12];
  const float* Wr_v     = (const float*)d_in[13];
  const float* W_O      = (const float*)d_in[14];
  float* out = (float*)d_out;

  char* ws = (char*)d_ws;
  size_t off = 0;
  auto take = [&](size_t bytes) { char* p = ws + off; off = (off + bytes + 255) & ~(size_t)255; return p; };
  bf16*  xb    = (bf16*) take((size_t)TOK * Dc * 2);
  bf16*  fbt   = (bf16*) take((size_t)Dc * Dc * 2);
  float* allh  = (float*)take((size_t)TOK * Dc * 4);
  bf16*  rqk_t = (bf16*) take((size_t)Dc * CDIM * 2);
  bf16*  rv_t  = (bf16*) take((size_t)Dc * CDIM * 2);
  bf16*  wo_t  = (bf16*) take((size_t)Dc * Dc * 2);
  bf16*  Cq    = (bf16*) take((size_t)TOK * CDIM * 2);
  bf16*  Ck    = (bf16*) take((size_t)TOK * CDIM * 2);
  bf16*  Cv    = (bf16*) take((size_t)TOK * CDIM * 2);
  float* Qf    = (float*)take((size_t)TOK * Dc * 4);
  float* Kf    = (float*)take((size_t)TOK * Dc * 4);
  float* Vf    = (float*)take((size_t)TOK * Dc * 4);
  bf16*  Qb    = (bf16*) take((size_t)TOK * Dc * 2);
  bf16*  Kb    = (bf16*) take((size_t)TOK * Dc * 2);
  bf16*  Vt    = (bf16*) take((size_t)TOK * Dc * 2);
  float* ao    = (float*)take((size_t)TOK * Dc * 4);
  bf16*  aso   = (bf16*) take((size_t)TOK * Dc * 2);
  float* qsc   = (float*)take((size_t)TOK * 4);
  float* macc  = (float*)take(48 * 4);

  ac_zero_acc<<<1, 64, 0, stream>>>(macc);
  ac_conv_x<<<(TOK * Dc + 255) / 256, 256, 0, stream>>>(x, xb);
  ac_conv_w<<<(2621440 + 255) / 256, 256, 0, stream>>>(f_neurons, r_neurons, W_O,
                                                       fbt, rqk_t, rv_t, wo_t);
  dim3 gemm_grid(Dc / 256, TOK / 32);
  ac_gemm_bf16<<<gemm_grid, 256, 0, stream>>>(xb, fbt, allh, TOK, Dc, Dc);
  ac_router<<<TOK, 256, 0, stream>>>(allh, fqk_wQ, fqk_wK, fv_w, fqk_emb, fv_emb,
                                     Wp_qk, bp_qk, Wp_v, bp_v, Wr_qk, Wr_v,
                                     Cq, Ck, Cv, macc);
  ac_gemm_bf16<<<gemm_grid, 256, 0, stream>>>(Cq, rqk_t, Qf, TOK, Dc, CDIM);
  ac_gemm_bf16<<<gemm_grid, 256, 0, stream>>>(Ck, rqk_t, Kf, TOK, Dc, CDIM);
  ac_gemm_bf16<<<gemm_grid, 256, 0, stream>>>(Cv, rv_t,  Vf, TOK, Dc, CDIM);
  ac_qscale<<<TOK, 256, 0, stream>>>(Qf, qsc);
  ac_pack_qkv<<<(TOK * Dc + 255) / 256, 256, 0, stream>>>(Qf, Kf, Vf, Qb, Kb, Vt);
  ac_attn<<<dim3(Sc / 16 / 2, Bc * Hc), 64, 0, stream>>>(Qb, Kb, Vt, ao);
  ac_scaleconv<<<(TOK * Dc + 255) / 256, 256, 0, stream>>>(ao, qsc, aso);
  ac_gemm_bf16<<<gemm_grid, 256, 0, stream>>>(aso, wo_t, out, TOK, Dc, Dc);
  ac_aux<<<1, 1, 0, stream>>>(macc, out);
}
